// SelfAttention_59356448031654
// MI455X (gfx1250) — matrix operand order repaired and checked by reading, not compiled, not run on hardware
//
#include <hip/hip_runtime.h>
#include <hip/hip_bf16.h>

// Self-attention (B=4, S=4096, D=256) for gfx1250 / MI455X.
// k1: fused QKV projection (f32 -> f16 fragments -> v_wmma_f32_16x16x32_f16).
//     q,k stored row-major f16; v stored transposed (vT[d][s]) for P@V B-frags.
// k2: flash attention, 16 queries/wave, 32 keys/step. K/V tiles are staged
//     cooperatively into double-buffered LDS via GLOBAL_LOAD_ASYNC_TO_LDS_B128
//     (ASYNCcnt + s_wait_asynccnt + s_barrier), overlapped with WMMA compute.
//     P is transposed C-layout -> A-layout through a per-wave LDS scratch.

typedef _Float16 half_t;
typedef __attribute__((ext_vector_type(16))) _Float16 v16h;
typedef __attribute__((ext_vector_type(8)))  _Float16 v8h;
typedef __attribute__((ext_vector_type(8)))  float    v8f;
typedef __attribute__((ext_vector_type(4)))  float    v4f;

#define KD   256
#define SEQ  4096
#define NBAT 4

// Padded LDS row strides (halves) -> conflict-free ds_load_b128 fragment reads
#define KSTRIDE 264   // 32 key rows x 264 halves  (528 B rows, 16B aligned)
#define VSTRIDE 40    // 256 d rows  x 40 halves   (80 B rows, 16B aligned)

#if defined(__has_builtin)
#if __has_builtin(__builtin_amdgcn_global_load_async_to_lds_b128) && \
    __has_builtin(__builtin_amdgcn_s_wait_asynccnt)
#define HAVE_ASYNC 1
#endif
#endif
#ifndef HAVE_ASYNC
#define HAVE_ASYNC 0
#endif

// Exact parameter type of the async-load builtin per hipcc diagnostic:
// pointer to GCC-style int vector_size(16), in AS1 (global) / AS3 (LDS).
typedef int i32x4_gcc __attribute__((vector_size(16)));
typedef __attribute__((address_space(1))) i32x4_gcc* gptr_b128;
typedef __attribute__((address_space(3))) i32x4_gcc* lptr_b128;

__device__ __forceinline__ v8f zero8() {
  v8f z;
  #pragma unroll
  for (int i = 0; i < 8; ++i) z[i] = 0.0f;
  return z;
}

__device__ __forceinline__ v16h mk16(v8h lo, v8h hi) {
  v16h a;
  #pragma unroll
  for (int i = 0; i < 8; ++i) { a[i] = lo[i]; a[8 + i] = hi[i]; }
  return a;
}

// 16-byte global -> LDS copy (async on gfx1250 toolchains that expose it)
__device__ __forceinline__ void copy16_g2l(const half_t* g, half_t* l) {
#if HAVE_ASYNC
  __builtin_amdgcn_global_load_async_to_lds_b128(
      (gptr_b128)(uintptr_t)g,
      (lptr_b128)(uint32_t)(uintptr_t)l,
      0, 0);
#else
  *(v8h*)l = *(const v8h*)g;
#endif
}

__device__ __forceinline__ void async_drain() {
#if HAVE_ASYNC
  __builtin_amdgcn_s_wait_asynccnt(0);
#endif
}

// A-matrix 16x32 f16 fragment from an fp32 row pointer.
// lane group g: slots 0-7 -> k = 32c+8g+h ; slots 8-15 -> k = 32c+16+8g+(h-8)
__device__ __forceinline__ v16h load_a_frag_f32(const float* __restrict__ rowp, int c, int g) {
  const float* p0 = rowp + 32 * c + 8 * g;
  const float* p1 = rowp + 32 * c + 16 + 8 * g;
  v4f f0 = *(const v4f*)(p0);
  v4f f1 = *(const v4f*)(p0 + 4);
  v4f f2 = *(const v4f*)(p1);
  v4f f3 = *(const v4f*)(p1 + 4);
  v16h a;
  #pragma unroll
  for (int i = 0; i < 4; ++i) {
    a[i]      = (half_t)f0[i];
    a[4 + i]  = (half_t)f1[i];
    a[8 + i]  = (half_t)f2[i];
    a[12 + i] = (half_t)f3[i];
  }
  return a;
}

// B-matrix 32x16 f16 fragment from an fp32 row pointer (row = output feature n).
// lane group g: slots 0-7 -> k = 32c+8g+h ; slots 8-15 -> k = 32c+16+8g+(h-8)
__device__ __forceinline__ v16h load_b_frag_f32(const float* __restrict__ rowp, int c, int g) {
  const float* p = rowp + 32 * c + 8 * g;
  v4f f0 = *(const v4f*)(p);
  v4f f1 = *(const v4f*)(p + 4);
  v4f f2 = *(const v4f*)(p + 16);
  v4f f3 = *(const v4f*)(p + 20);
  v16h b;
  #pragma unroll
  for (int i = 0; i < 4; ++i) {
    b[i]      = (half_t)f0[i];
    b[4 + i]  = (half_t)f1[i];
    b[8 + i]  = (half_t)f2[i];
    b[12 + i] = (half_t)f3[i];
  }
  return b;
}

__global__ __launch_bounds__(256)
void qkv_kernel(const float* __restrict__ x,
                const float* __restrict__ Wq, const float* __restrict__ bq,
                const float* __restrict__ Wk, const float* __restrict__ bk,
                const float* __restrict__ Wv, const float* __restrict__ bv,
                half_t* __restrict__ qh, half_t* __restrict__ kh,
                half_t* __restrict__ vT)
{
  const int wave = threadIdx.x >> 5;
  const int lane = threadIdx.x & 31;
  const int g  = lane >> 4;
  const int lm = lane & 15;
  const int gw   = blockIdx.x * 8 + wave;   // 0..1023 wave tiles
  const int row0 = gw * 16;                 // flat row base in [0, B*S)
  const int b    = row0 / SEQ;
  const int s0   = row0 % SEQ;

  // Load all 8 A-fragments (this wave's 16 rows of x) once.
  const float* xrow = x + (size_t)(row0 + lm) * KD;
  v16h A[8];
  #pragma unroll
  for (int c = 0; c < 8; ++c) A[c] = load_a_frag_f32(xrow, c, g);

  const float* Wmat[3] = {Wq, Wk, Wv};
  const float* Bias[3] = {bq, bk, bv};

  #pragma unroll
  for (int mat = 0; mat < 3; ++mat) {
    const float* W    = Wmat[mat];
    const float* bias = Bias[mat];
    for (int nt = 0; nt < 16; ++nt) {
      const int n0 = nt * 16;
      const float* wrow = W + (size_t)(n0 + lm) * KD;  // y = x @ W^T -> B[k][n] = W[n][k]
      v8f acc = zero8();
      #pragma unroll
      for (int c = 0; c < 8; ++c) {
        v16h Bf = load_b_frag_f32(wrow, c, g);
        acc = __builtin_amdgcn_wmma_f32_16x16x32_f16(
            false, A[c], false, Bf, (short)0, acc, false, false);
      }
      const float bval = bias[n0 + lm];
      if (mat == 0) {
        // q scaled by 1/sqrt(D) = 1/16, row-major f16
        #pragma unroll
        for (int r = 0; r < 8; ++r) {
          float v = (acc[r] + bval) * 0.0625f;
          qh[(size_t)(row0 + 8 * g + r) * KD + n0 + lm] = (half_t)v;
        }
      } else if (mat == 1) {
        #pragma unroll
        for (int r = 0; r < 8; ++r) {
          kh[(size_t)(row0 + 8 * g + r) * KD + n0 + lm] = (half_t)(acc[r] + bval);
        }
      } else {
        // v transposed: vT[b][d][s]; C-layout rows r+8g are consecutive s -> b128 store
        v8h pack;
        #pragma unroll
        for (int r = 0; r < 8; ++r) pack[r] = (half_t)(acc[r] + bval);
        half_t* dst = vT + ((size_t)b * KD + n0 + lm) * SEQ + s0 + 8 * g;
        *(v8h*)dst = pack;
      }
    }
  }
}

// Stage one 32-key step: K tile (32 x 256 halves) and V tile (256 x 32 halves)
// into padded LDS buffers. 2048 16-byte chunks spread over 256 threads,
// consecutive lanes -> consecutive 16B chunks (coalesced).
__device__ __forceinline__ void stage_kv(const half_t* __restrict__ kb,
                                         const half_t* __restrict__ vb,
                                         int j0, half_t* ldsK, half_t* ldsV, int tid) {
  #pragma unroll
  for (int i = 0; i < 4; ++i) {
    const int c  = tid + 256 * i;        // 0..1023
    const int kr = c >> 5;               // key row 0..31
    const int kc = (c & 31) * 8;         // d col in halves
    copy16_g2l(kb + (size_t)(j0 + kr) * KD + kc, ldsK + kr * KSTRIDE + kc);
    const int vr = c >> 2;               // d row 0..255
    const int vc = (c & 3) * 8;          // key col in halves
    copy16_g2l(vb + (size_t)vr * SEQ + j0 + vc, ldsV + vr * VSTRIDE + vc);
  }
}

__global__ __launch_bounds__(256)
void attn_kernel(const half_t* __restrict__ qh, const half_t* __restrict__ kh,
                 const half_t* __restrict__ vT, float* __restrict__ out)
{
  __shared__ __align__(16) half_t ldsK[2 * 32 * KSTRIDE];   // 33792 B
  __shared__ __align__(16) half_t ldsV[2 * 256 * VSTRIDE];  // 40960 B
  __shared__ __align__(16) half_t ldsP[8 * 16 * 40];        // 10240 B

  const int tid  = threadIdx.x;
  const int wave = tid >> 5;
  const int lane = tid & 31;
  const int g  = lane >> 4;
  const int lm = lane & 15;
  const int gw = blockIdx.x * 8 + wave;     // 0..1023 (8 query tiles per block, same batch)
  const int b  = gw >> 8;                   // 256 query tiles per batch
  const int s0 = (gw & 255) * 16;

  const half_t* qb = qh + (size_t)b * SEQ * KD;
  const half_t* kb = kh + (size_t)b * SEQ * KD;
  const half_t* vb = vT + (size_t)b * KD * SEQ;
  float*        ob = out + (size_t)b * SEQ * KD;

  // Kick off the first K/V tile DMA before anything else.
  stage_kv(kb, vb, 0, ldsK, ldsV, tid);

  // Q A-fragments (pre-scaled by 1/16 in qkv_kernel), f16 row-major (overlaps DMA)
  v16h QA[8];
  {
    const half_t* qrow = qb + (size_t)(s0 + lm) * KD;
    #pragma unroll
    for (int c = 0; c < 8; ++c) {
      v8h lo = *(const v8h*)(qrow + 32 * c + 8 * g);
      v8h hi = *(const v8h*)(qrow + 32 * c + 16 + 8 * g);
      QA[c] = mk16(lo, hi);
    }
  }

  v8f O[16];
  #pragma unroll
  for (int t = 0; t < 16; ++t) O[t] = zero8();

  float mstat[8], lstat[8];
  #pragma unroll
  for (int r = 0; r < 8; ++r) { mstat[r] = -1.0e30f; lstat[r] = 0.0f; }

  half_t* myp = &ldsP[wave * 16 * 40];

  async_drain();     // own slice of tile 0 complete
  __syncthreads();   // all slices visible

  for (int j0 = 0; j0 < SEQ; j0 += 32) {
    const int cur = (j0 >> 5) & 1;
    half_t* ldsKc = ldsK + cur * (32 * KSTRIDE);
    half_t* ldsVc = ldsV + cur * (256 * VSTRIDE);

    // Issue DMA for the next tile into the other buffer (overlapped with WMMA).
    if (j0 + 32 < SEQ) {
      stage_kv(kb, vb, j0 + 32,
               ldsK + (1 - cur) * (32 * KSTRIDE),
               ldsV + (1 - cur) * (256 * VSTRIDE), tid);
    }

    // scores: two 16x16 tiles covering keys [j0, j0+32), B-frags from LDS
    v8f S0 = zero8(), S1 = zero8();
    const half_t* kr0 = ldsKc + lm * KSTRIDE;          // key = j0 + n
    const half_t* kr1 = ldsKc + (16 + lm) * KSTRIDE;   // key = j0 + 16 + n
    #pragma unroll
    for (int c = 0; c < 8; ++c) {
      const int off = 32 * c + 8 * g;
      v16h B0 = mk16(*(const v8h*)(kr0 + off), *(const v8h*)(kr0 + off + 16));
      S0 = __builtin_amdgcn_wmma_f32_16x16x32_f16(
          false, QA[c], false, B0, (short)0, S0, false, false);
      v16h B1 = mk16(*(const v8h*)(kr1 + off), *(const v8h*)(kr1 + off + 16));
      S1 = __builtin_amdgcn_wmma_f32_16x16x32_f16(
          false, QA[c], false, B1, (short)0, S1, false, false);
    }

    // online softmax; row stats reduced across the 16-lane half-group
    #pragma unroll
    for (int r = 0; r < 8; ++r) {
      float t = fmaxf(S0[r], S1[r]);
      t = fmaxf(t, __shfl_xor(t, 1, 32));
      t = fmaxf(t, __shfl_xor(t, 2, 32));
      t = fmaxf(t, __shfl_xor(t, 4, 32));
      t = fmaxf(t, __shfl_xor(t, 8, 32));
      const float mnew  = fmaxf(mstat[r], t);
      const float alpha = __expf(mstat[r] - mnew);
      mstat[r] = mnew;
      const float p0 = __expf(S0[r] - mnew);
      const float p1 = __expf(S1[r] - mnew);
      float rs = p0 + p1;
      rs += __shfl_xor(rs, 1, 32);
      rs += __shfl_xor(rs, 2, 32);
      rs += __shfl_xor(rs, 4, 32);
      rs += __shfl_xor(rs, 8, 32);
      lstat[r] = lstat[r] * alpha + rs;
      #pragma unroll
      for (int tt = 0; tt < 16; ++tt) O[tt][r] *= alpha;
      // write P (C-layout: row = r+8g, cols lm / 16+lm) to per-wave LDS scratch
      const int row = 8 * g + r;
      myp[row * 40 + lm]      = (half_t)p0;
      myp[row * 40 + 16 + lm] = (half_t)p1;
    }

    // read P back as an A-matrix 16x32 fragment (same-wave DScnt ordering)
    v16h PA = mk16(*(const v8h*)(myp + lm * 40 + 8 * g),
                   *(const v8h*)(myp + lm * 40 + 16 + 8 * g));

    // O += P @ V : B[k][n] = vT[n0+n][j0+k], served from padded LDS V tile
    #pragma unroll
    for (int t = 0; t < 16; ++t) {
      const half_t* vrow = ldsVc + (16 * t + lm) * VSTRIDE + 8 * g;
      v16h Bv = mk16(*(const v8h*)(vrow), *(const v8h*)(vrow + 16));
      O[t] = __builtin_amdgcn_wmma_f32_16x16x32_f16(
          false, PA, false, Bv, (short)0, O[t], false, false);
    }

    async_drain();    // own next-tile DMA complete
    __syncthreads();  // everyone done reading cur / writing next
  }

  // finalize: divide by softmax denominator, store fp32 out[b][s][d]
  float rl[8];
  #pragma unroll
  for (int r = 0; r < 8; ++r) rl[r] = 1.0f / lstat[r];
  #pragma unroll
  for (int t = 0; t < 16; ++t) {
    #pragma unroll
    for (int r = 0; r < 8; ++r) {
      ob[(size_t)(s0 + 8 * g + r) * KD + 16 * t + lm] = O[t][r] * rl[r];
    }
  }
}

extern "C" void kernel_launch(void* const* d_in, const int* in_sizes, int n_in,
                              void* d_out, int out_size, void* d_ws, size_t ws_size,
                              hipStream_t stream) {
  const float* x  = (const float*)d_in[0];
  const float* Wq = (const float*)d_in[1];
  const float* bq = (const float*)d_in[2];
  const float* Wk = (const float*)d_in[3];
  const float* bk = (const float*)d_in[4];
  const float* Wv = (const float*)d_in[5];
  const float* bv = (const float*)d_in[6];
  float* out = (float*)d_out;

  char* ws = (char*)d_ws;
  const size_t qbytes = (size_t)NBAT * SEQ * KD * sizeof(half_t);  // 8 MB each
  half_t* qh = (half_t*)(ws);
  half_t* kh = (half_t*)(ws + qbytes);
  half_t* vT = (half_t*)(ws + 2 * qbytes);

  // 1024 wave-tiles of 16 rows each, 8 waves per 256-thread block
  qkv_kernel<<<128, 256, 0, stream>>>(x, Wq, bq, Wk, bk, Wv, bv, qh, kh, vT);
  attn_kernel<<<128, 256, 0, stream>>>(qh, kh, vT, out);
}
